// BlockwiseAttention3D_77017353551958
// MI455X (gfx1250) — matrix operand order repaired and checked by reading, not compiled, not run on hardware
//
#include <hip/hip_runtime.h>

// CDNA5 WMMA vector operand types
typedef __attribute__((ext_vector_type(16))) _Float16 v16h;
typedef __attribute__((ext_vector_type(8)))  float    v8f;
typedef __attribute__((ext_vector_type(4)))  float    v4f;

#define NH  2      // heads
#define BS  8      // block size
#define L   512    // BS^3 tokens per block
#define NB  6      // 48 / BS blocks per axis
#define DIM 48

// One workgroup per 8x8x8 spatial block; 512 threads = 16 wave32 waves.
// Each wave owns TWO 16-row tiles (tile=wave and tile=wave+16) of the
// 512x512 attention matrix and processes both heads, sharing the k-loads
// and the B matrix between the two tiles' WMMAs.
__global__ __launch_bounds__(512)
void blockattn3d_wmma_kernel(const float* __restrict__ x,
                             const float* __restrict__ wq, const float* __restrict__ bq,
                             const float* __restrict__ wk, const float* __restrict__ bk,
                             const float* __restrict__ wv, const float* __restrict__ bv,
                             float* __restrict__ out)
{
    __shared__ __align__(16) float qsL[NH][L];
    __shared__ __align__(16) float ksL[NH][L];
    __shared__ __align__(16) float vsL[NH][L];
    __shared__ float redmx[512];
    __shared__ float redmn[512];
    __shared__ float kext[NH][2];   // [h][0]=min(k), [h][1]=max(k)

    const int tid  = threadIdx.x;
    const int lane = tid & 31;
    const int wave = tid >> 5;

    const int blk = blockIdx.x;
    const int bw_ = blk % NB;
    const int bh_ = (blk / NB) % NB;
    const int bd_ = blk / (NB * NB);

    // ---- 1x1x1 conv (C=1): q/k/v = x*w[h] + b[h], straight into LDS ----
    {
        const int l  = tid;                      // blockDim.x == L
        const int z  = l >> 6, y = (l >> 3) & 7, xx = l & 7;
        const int gi = ((bd_ * BS + z) * DIM + (bh_ * BS + y)) * DIM + (bw_ * BS + xx);
        const float xv = x[gi];
#pragma unroll
        for (int h = 0; h < NH; ++h) {
            qsL[h][l] = fmaf(xv, wq[h], bq[h]);
            ksL[h][l] = fmaf(xv, wk[h], bk[h]);
            vsL[h][l] = fmaf(xv, wv[h], bv[h]);
        }
    }
    __syncthreads();

    // ---- per-head min/max of k: exact row-max of rank-1 scores is
    //      q_m * (q_m>=0 ? kmax : kmin), used for softmax stability ----
    for (int h = 0; h < NH; ++h) {
        const float kv = ksL[h][tid];
        redmx[tid] = kv;
        redmn[tid] = kv;
        __syncthreads();
        for (int s = 256; s > 0; s >>= 1) {
            if (tid < s) {
                redmx[tid] = fmaxf(redmx[tid], redmx[tid + s]);
                redmn[tid] = fminf(redmn[tid], redmn[tid + s]);
            }
            __syncthreads();
        }
        if (tid == 0) { kext[h][0] = redmn[0]; kext[h][1] = redmx[0]; }
        __syncthreads();
    }

    // ---- attention ----
    // B = [ v | 1 | 0 ... 0 ]  =>  D[:,0] = sum(exp*v), D[:,1] = sum(exp)
    const float LOG2E = 1.44269504088896340736f;
    const float sc    = LOG2E / (float)L;   // fold log2(e): v_exp_f32 is 2^x
    const int   ncol  = lane & 15;          // D/B column owned by this lane
    const bool  hi    = lane >= 16;
    const int   kOffA = hi ? 8  : 0;        // A layout: high lanes hold K+8
    const int   kOffB = hi ? 16 : 0;        // unused: B is loaded with kOffA (K+8, same as A)
    const float msk0  = (ncol == 0) ? 1.0f : 0.0f;   // branchless B build
    const float msk1  = (ncol == 1) ? 1.0f : 0.0f;

    const int mA = wave * 16 + ncol;        // my row in tile `wave`
    const int mB = mA + 256;                // my row in tile `wave+16`

    float obA = 0.0f, obB = 0.0f;           // head-summed outputs (lanes 0-15)

#pragma unroll
    for (int h = 0; h < NH; ++h) {
        const float qa = qsL[h][mA];
        const float qb = qsL[h][mB];
        const float sa = qa * sc;
        const float sb = qb * sc;
        const float ca = sa * (qa >= 0.0f ? kext[h][1] : kext[h][0]);
        const float cb = sb * (qb >= 0.0f ? kext[h][1] : kext[h][0]);

        v8f acc0 = {};
        v8f acc1 = {};
        for (int kc = 0; kc < 16; ++kc) {
            const int n0 = kc * 32;

            // k values for this lane: two contiguous runs of 8 -> 4x ds_load_b128
            const v4f* kp0 = (const v4f*)&ksL[h][n0 + kOffA];
            const v4f* kp1 = (const v4f*)&ksL[h][n0 + 16 + kOffA];
            const v4f  k0 = kp0[0], k1 = kp0[1], k2 = kp1[0], k3 = kp1[1];
            // v values for this lane: two contiguous runs of 8 (same k order as A) -> 4x ds_load_b128
            const v4f* vp0 = (const v4f*)&vsL[h][n0 + kOffA];
            const v4f* vp1 = (const v4f*)&vsL[h][n0 + 16 + kOffA];
            const v4f  w0 = vp0[0], w1 = vp0[1], w2 = vp1[0], w3 = vp1[1];

            float kv[16], vv[16];
#pragma unroll
            for (int j = 0; j < 4; ++j) {
                kv[j] = k0[j]; kv[4 + j] = k1[j]; kv[8 + j] = k2[j]; kv[12 + j] = k3[j];
                vv[j] = w0[j]; vv[4 + j] = w1[j]; vv[8 + j] = w2[j]; vv[12 + j] = w3[j];
            }

            v16h A0, A1, Bm;
#pragma unroll
            for (int e = 0; e < 16; ++e) {
                A0[e] = (_Float16)__builtin_amdgcn_exp2f(fmaf(sa, kv[e], -ca));
                A1[e] = (_Float16)__builtin_amdgcn_exp2f(fmaf(sb, kv[e], -cb));
                Bm[e] = (_Float16)fmaf(vv[e], msk0, msk1);   // col0=v, col1=1, else 0
            }

            acc0 = __builtin_amdgcn_wmma_f32_16x16x32_f16(
                       false, A0, false, Bm, (short)0, acc0, false, false);
            acc1 = __builtin_amdgcn_wmma_f32_16x16x32_f16(
                       false, A1, false, Bm, (short)0, acc1, false, false);
        }

        // Extract D[:,0] (num) and D[:,1] (den). Per 16x16 f32 C/D layout:
        // row r lives in VGPR r&7 of lane (col + (r>=8 ? 16 : 0)).
        const int srcN = (lane & 8) ? 16 : 0;
        const int srcD = srcN + 1;
        float numA = 0.0f, denA = 1.0f, numB = 0.0f, denB = 1.0f;
#pragma unroll
        for (int r = 0; r < 8; ++r) {
            const float nA = __shfl(acc0[r], srcN, 32);
            const float dA = __shfl(acc0[r], srcD, 32);
            const float nB = __shfl(acc1[r], srcN, 32);
            const float dB = __shfl(acc1[r], srcD, 32);
            const bool sel = ((lane & 7) == r);
            numA = sel ? nA : numA;  denA = sel ? dA : denA;
            numB = sel ? nB : numB;  denB = sel ? dB : denB;
        }
        obA += numA / denA;
        obB += numB / denB;
    }

    if (lane < 16) {
        int z = mA >> 6, y = (mA >> 3) & 7, xx = mA & 7;
        out[((bd_ * BS + z) * DIM + (bh_ * BS + y)) * DIM + (bw_ * BS + xx)] = obA;
        z = mB >> 6; y = (mB >> 3) & 7; xx = mB & 7;
        out[((bd_ * BS + z) * DIM + (bh_ * BS + y)) * DIM + (bw_ * BS + xx)] = obB;
    }
}

extern "C" void kernel_launch(void* const* d_in, const int* in_sizes, int n_in,
                              void* d_out, int out_size, void* d_ws, size_t ws_size,
                              hipStream_t stream) {
    (void)in_sizes; (void)n_in; (void)out_size; (void)d_ws; (void)ws_size;
    const float* x  = (const float*)d_in[0];
    const float* wq = (const float*)d_in[1];
    const float* bq = (const float*)d_in[2];
    const float* wk = (const float*)d_in[3];
    const float* bk = (const float*)d_in[4];
    const float* wv = (const float*)d_in[5];
    const float* bv = (const float*)d_in[6];
    float* out = (float*)d_out;

    dim3 grid(NB * NB * NB);   // 216 spatial blocks
    dim3 block(512);           // 16 wave32 waves
    blockattn3d_wmma_kernel<<<grid, block, 0, stream>>>(x, wq, bq, wk, bk, wv, bv, out);
}
